// OCAB_59940563583776
// MI455X (gfx1250) — matrix operand order repaired and checked by reading, not compiled, not run on hardware
//
#include <hip/hip_runtime.h>
#include <hip/hip_bf16.h>

typedef __attribute__((ext_vector_type(16))) _Float16 v16h;
typedef __attribute__((ext_vector_type(8)))  _Float16 v8h;
typedef __attribute__((ext_vector_type(8)))  float    v8f;
typedef __attribute__((ext_vector_type(4)))  float    v4f;
typedef __attribute__((ext_vector_type(4)))  int      v4i;
typedef __attribute__((address_space(1)))    v4i      as1_v4i;
typedef __attribute__((address_space(3)))    v4i      as3_v4i;

#define NPIX   131072      // B*T*H*W = 2*4*128*128
#define CDIM   192
#define HEADS  6
#define HD     32          // head dim
#define NWIN   2048        // B*T*16*16
#define MLPH   384
#define KPAD   160         // 144 overlapping keys padded to 160

#if __has_builtin(__builtin_amdgcn_global_load_async_to_lds_b128)
#define HAVE_ASYNC_LDS 1
#else
#define HAVE_ASYNC_LDS 0
#endif

// ---------------------------------------------------------------------------
// 16-byte global -> LDS copy; async (ASYNCcnt-tracked) when available.
// ---------------------------------------------------------------------------
__device__ __forceinline__ void copy16_g2l(const _Float16* gsrc, _Float16* ldst) {
#if HAVE_ASYNC_LDS
    __builtin_amdgcn_global_load_async_to_lds_b128(
        (as1_v4i*)(uintptr_t)gsrc,
        (as3_v4i*)(uint32_t)(uintptr_t)ldst, 0, 0);
#else
    *(v8h*)ldst = *(const v8h*)gsrc;
#endif
}

__device__ __forceinline__ void wait_async_copies() {
#if HAVE_ASYNC_LDS
#if __has_builtin(__builtin_amdgcn_s_wait_asynccnt)
    __builtin_amdgcn_s_wait_asynccnt(0);
#else
    asm volatile("s_wait_asynccnt 0" ::: "memory");
#endif
#endif
}

// ---------------------------------------------------------------------------
// WMMA fragment loaders (row-major f16 source, ld in elements).
// A (16x32, MxK): lanes 0-15 row M=lane, k [k0..k0+15];
//                 lanes 16-31 row M=lane-16, k [k0+16..k0+31].
// B (32x16, KxN) loaded from TRANSPOSED storage Bt[N][K]:
//                 lanes 0-15 col N=lane, k [k0..k0+15];
//                 lanes 16-31 col N=lane-16, k [k0+16..k0+31].
// ---------------------------------------------------------------------------
__device__ __forceinline__ v16h load_fragA(const _Float16* base, int row0, int k0, int ld) {
    int lane = threadIdx.x & 31;
    int hh = lane >> 4;
    int r  = lane & 15;
    const _Float16* p = base + (size_t)(row0 + r) * ld + k0 + hh * 16;
    v8h lo = *(const v8h*)p;
    v8h hi = *(const v8h*)(p + 8);
    v16h a;
#pragma unroll
    for (int i = 0; i < 8; ++i) { a[i] = lo[i]; a[8 + i] = hi[i]; }
    return a;
}

__device__ __forceinline__ v16h load_fragB(const _Float16* Bt, int n0, int k0, int ld) {
    int lane = threadIdx.x & 31;
    int hh = lane >> 4;
    int n  = lane & 15;
    const _Float16* p = Bt + (size_t)(n0 + n) * ld + k0 + hh * 16;
    return *(const v16h*)p;
}

__device__ __forceinline__ v8f wmma_f16(v16h a, v16h b, v8f c) {
    return __builtin_amdgcn_wmma_f32_16x16x32_f16(false, a, false, b, (short)0, c, false, false);
}

// ---------------------------------------------------------------------------
// Kernel 1: weight transpose+cast to f16, bias-table gather.
// Bias is padded to [HEADS][64][KPAD] and pre-divided by the attention scale
// so it can seed the QK accumulator before the WMMA (scale applied once after).
// ---------------------------------------------------------------------------
__global__ void prep_kernel(const float* __restrict__ qkv_w, const float* __restrict__ fc1_w,
                            const float* __restrict__ fc2_w, const float* __restrict__ rpb,
                            const int* __restrict__ rpi,
                            _Float16* qkvwT, _Float16* fc1wT, _Float16* fc2wT, float* biasP) {
    const float inv_scale = 5.656854249492380f;  // sqrt(32) = 1/scale
    const int total = 192 * 576 + 192 * 384 + 384 * 192 + HEADS * 64 * KPAD;
    for (int i = blockIdx.x * blockDim.x + threadIdx.x; i < total; i += gridDim.x * blockDim.x) {
        int t = i;
        if (t < 192 * 576) { int r = t / 576, c = t % 576; qkvwT[(size_t)c * 192 + r] = (_Float16)qkv_w[t]; continue; }
        t -= 192 * 576;
        if (t < 192 * 384) { int r = t / 384, c = t % 384; fc1wT[(size_t)c * 192 + r] = (_Float16)fc1_w[t]; continue; }
        t -= 192 * 384;
        if (t < 384 * 192) { int r = t / 192, c = t % 192; fc2wT[(size_t)c * 384 + r] = (_Float16)fc2_w[t]; continue; }
        t -= 384 * 192;
        int h = t / (64 * KPAD), rem = t % (64 * KPAD);
        int q = rem / KPAD, k = rem % KPAD;
        biasP[t] = (k < 144) ? rpb[(size_t)rpi[q * 144 + k] * HEADS + h] * inv_scale : 0.0f;
    }
}

// ---------------------------------------------------------------------------
// LayerNorm: one wave (32 lanes) per 192-element row, f16 output.
// ---------------------------------------------------------------------------
__global__ __launch_bounds__(256) void ln_kernel(const float* __restrict__ x,
                                                 const float* __restrict__ g,
                                                 const float* __restrict__ b,
                                                 _Float16* __restrict__ out, int nrows) {
    int row = blockIdx.x * 8 + (threadIdx.x >> 5);
    if (row >= nrows) return;
    int lane = threadIdx.x & 31;
    const float* xr = x + (size_t)row * CDIM;
    float v[6], s = 0.f, s2 = 0.f;
#pragma unroll
    for (int j = 0; j < 6; ++j) { v[j] = xr[lane + j * 32]; s += v[j]; s2 += v[j] * v[j]; }
#pragma unroll
    for (int off = 16; off > 0; off >>= 1) { s += __shfl_xor(s, off, 32); s2 += __shfl_xor(s2, off, 32); }
    float mean = s * (1.f / CDIM);
    float var  = s2 * (1.f / CDIM) - mean * mean;
    float rstd = rsqrtf(var + 1e-5f);
    _Float16* o = out + (size_t)row * CDIM;
#pragma unroll
    for (int j = 0; j < 6; ++j) {
        int idx = lane + j * 32;
        o[idx] = (_Float16)((v[j] - mean) * rstd * g[idx] + b[idx]);
    }
}

// ---------------------------------------------------------------------------
// Generic WMMA GEMM: C[M,N] = A[M,K] @ Bt[N,K]^T + bias, wave tile 16x64.
// MODE 0: f16 out.  MODE 1: exact GELU -> f16 out.  MODE 2: + residual -> f32.
// Epilogue stages the tile through LDS so global stores are coalesced b128.
// ---------------------------------------------------------------------------
template <int MODE>
__global__ __launch_bounds__(256) void gemm_kernel(const _Float16* __restrict__ A,
                                                   const _Float16* __restrict__ Bt,
                                                   const float* __restrict__ bias,
                                                   _Float16* __restrict__ outH,
                                                   float* __restrict__ outF,
                                                   const float* __restrict__ residual,
                                                   int M, int N, int K) {
    __shared__ __align__(16) char smem[8 * 16 * 64 * ((MODE == 2) ? 4 : 2)];

    int wave = (blockIdx.x * blockDim.x + threadIdx.x) >> 5;
    int tiles_n = N >> 6;
    int tm = wave / tiles_n, tn = wave % tiles_n;
    if (tm >= (M >> 4)) return;
    int m0 = tm << 4, n0 = tn << 6;

    v8f acc[4] = {};
    for (int k0 = 0; k0 < K; k0 += 32) {
        v16h a = load_fragA(A, m0, k0, K);
#pragma unroll
        for (int j = 0; j < 4; ++j) {
            v16h b = load_fragB(Bt, n0 + 16 * j, k0, K);
            acc[j] = wmma_f16(a, b, acc[j]);
        }
    }

    int wslot = (threadIdx.x >> 5) & 7;
    int lane = threadIdx.x & 31;
    int n = lane & 15, mh = (lane >> 4) * 8;

    if constexpr (MODE == 2) {
        float* tile = (float*)smem + wslot * (16 * 64);
#pragma unroll
        for (int j = 0; j < 4; ++j) {
            float bv = bias[n0 + j * 16 + n];
#pragma unroll
            for (int r = 0; r < 8; ++r)
                tile[(mh + r) * 64 + j * 16 + n] = acc[j][r] + bv;
        }
        // coalesced b128 stores + fused residual
#pragma unroll
        for (int i = 0; i < 8; ++i) {
            int c = i * 32 + lane;                 // 0..255 chunks of 4 floats
            int row = c >> 4, cc = (c & 15) * 4;
            v4f v = *(const v4f*)(tile + row * 64 + cc);
            size_t gidx = (size_t)(m0 + row) * N + n0 + cc;
            v4f rsd = *(const v4f*)(residual + gidx);
            *(v4f*)(outF + gidx) = v + rsd;
        }
    } else {
        _Float16* tile = (_Float16*)smem + wslot * (16 * 64);
#pragma unroll
        for (int j = 0; j < 4; ++j) {
            float bv = bias[n0 + j * 16 + n];
#pragma unroll
            for (int r = 0; r < 8; ++r) {
                float val = acc[j][r] + bv;
                if (MODE == 1) val = 0.5f * val * (1.0f + erff(val * 0.70710678118654752f));
                tile[(mh + r) * 64 + j * 16 + n] = (_Float16)val;
            }
        }
        // coalesced b128 stores
#pragma unroll
        for (int i = 0; i < 4; ++i) {
            int c = i * 32 + lane;                 // 0..127 chunks of 8 halfs
            int row = c >> 3, cc = (c & 7) * 8;
            v8h v = *(const v8h*)(tile + row * 64 + cc);
            *(v8h*)(outH + (size_t)(m0 + row) * N + n0 + cc) = v;
        }
    }
}

// ---------------------------------------------------------------------------
// Attention: one block (4 waves, 128 thr) per (window, head).
// 64 queries x 144 overlapping keys (padded to 160), bias seeds the QK
// accumulator (pre-divided by scale), no softmax.  x1 = attn_out + shortcut.
// ---------------------------------------------------------------------------
__global__ __launch_bounds__(128) void attn_kernel(const _Float16* __restrict__ qkv,
                                                   const float* __restrict__ x,
                                                   const float* __restrict__ biasP,
                                                   float* __restrict__ x1) {
    int head = blockIdx.x % HEADS;
    int win  = blockIdx.x / HEADS;
    int ww = win & 15, wh = (win >> 4) & 15, bt = win >> 8;
    const size_t planeBase = (size_t)bt * (128 * 128);
    const float scale = 0.17677669529663687f;  // 1/sqrt(32)

    __shared__ __align__(16) _Float16 sQ[64 * 32];
    __shared__ __align__(16) _Float16 sK[KPAD * 32];
    __shared__ __align__(16) _Float16 sVT[32 * KPAD];
    __shared__ __align__(16) _Float16 sAttn[64 * KPAD];

    int tid = threadIdx.x;

    // Q tile (raw; scale applied after the QK WMMA) — async copy to LDS
    for (int c = tid; c < 64 * 4; c += 128) {
        int row = c >> 2, c8 = (c & 3) * 8;
        int qy = row >> 3, qx = row & 7;
        size_t pix = planeBase + (size_t)(wh * 8 + qy) * 128 + (ww * 8 + qx);
        copy16_g2l(qkv + pix * 576 + head * HD + c8, &sQ[row * 32 + c8]);
    }
    // K tile, rows 144..159 + out-of-image pixels zeroed
    for (int c = tid; c < KPAD * 4; c += 128) {
        int row = c >> 2, c8 = (c & 3) * 8;
        v8h v = {};
        if (row < 144) {
            int gy = wh * 8 - 2 + row / 12, gx = ww * 8 - 2 + row % 12;
            if (gy >= 0 && gy < 128 && gx >= 0 && gx < 128) {
                size_t pix = planeBase + (size_t)gy * 128 + gx;
                v = *(const v8h*)(qkv + pix * 576 + CDIM + head * HD + c8);
            }
        }
        *(v8h*)(&sK[row * 32 + c8]) = v;
    }
    // V tile, stored transposed (Vt[d][key]) for B-fragment loads
    for (int c = tid; c < KPAD * 4; c += 128) {
        int row = c >> 2, c8 = (c & 3) * 8;
        v8h v = {};
        if (row < 144) {
            int gy = wh * 8 - 2 + row / 12, gx = ww * 8 - 2 + row % 12;
            if (gy >= 0 && gy < 128 && gx >= 0 && gx < 128) {
                size_t pix = planeBase + (size_t)gy * 128 + gx;
                v = *(const v8h*)(qkv + pix * 576 + 2 * CDIM + head * HD + c8);
            }
        }
#pragma unroll
        for (int i = 0; i < 8; ++i) sVT[(c8 + i) * KPAD + row] = v[i];
    }
    wait_async_copies();
    __syncthreads();

    int wv = tid >> 5, lane = tid & 31;
    int m0 = wv * 16;
    int nn = lane & 15, mh = (lane >> 4) * 8;

    // attn = scale * (q @ k^T) + bias : accumulator seeded with bias/scale
    v16h aq = load_fragA(sQ, m0, 0, 32);
    for (int nt = 0; nt < 10; ++nt) {
        int ncol = nt * 16 + nn;
        const float* bp = biasP + ((size_t)head * 64 + m0 + mh) * KPAD + ncol;
        v8f acc;
#pragma unroll
        for (int r = 0; r < 8; ++r) acc[r] = bp[r * KPAD];
        v16h b = load_fragB(sK, nt * 16, 0, 32);
        acc = wmma_f16(aq, b, acc);
#pragma unroll
        for (int r = 0; r < 8; ++r)
            sAttn[(m0 + mh + r) * KPAD + ncol] = (_Float16)(acc[r] * scale);
    }

    // out = attn @ v   (K = 160, zero-padded tail exact)
    v8f acc2[2] = {};
    for (int k0 = 0; k0 < KPAD; k0 += 32) {
        v16h a2 = load_fragA(sAttn, m0, k0, KPAD);
#pragma unroll
        for (int j = 0; j < 2; ++j) {
            v16h b = load_fragB(sVT, j * 16, k0, KPAD);
            acc2[j] = wmma_f16(a2, b, acc2[j]);
        }
    }
#pragma unroll
    for (int j = 0; j < 2; ++j) {
        int col = head * HD + j * 16 + nn;
#pragma unroll
        for (int r = 0; r < 8; ++r) {
            int qrow = m0 + mh + r;
            int qy = qrow >> 3, qx = qrow & 7;
            size_t pix = planeBase + (size_t)(wh * 8 + qy) * 128 + (ww * 8 + qx);
            x1[pix * CDIM + col] = acc2[j][r] + x[pix * CDIM + col];
        }
    }
}

// ---------------------------------------------------------------------------
extern "C" void kernel_launch(void* const* d_in, const int* in_sizes, int n_in,
                              void* d_out, int out_size, void* d_ws, size_t ws_size,
                              hipStream_t stream) {
    const float* x    = (const float*)d_in[0];
    const int*   rpi  = (const int*)  d_in[2];
    const float* n1g  = (const float*)d_in[3];
    const float* n1b  = (const float*)d_in[4];
    const float* qkvw = (const float*)d_in[5];
    const float* qkvb = (const float*)d_in[6];
    const float* rpb  = (const float*)d_in[7];
    const float* n2g  = (const float*)d_in[8];
    const float* n2b  = (const float*)d_in[9];
    const float* fc1w = (const float*)d_in[10];
    const float* fc1b = (const float*)d_in[11];
    const float* fc2w = (const float*)d_in[12];
    const float* fc2b = (const float*)d_in[13];

    auto align256 = [](size_t v) { return (v + 255) & ~(size_t)255; };
    char* ws = (char*)d_ws;
    size_t off = 0;
    _Float16* qkvwT = (_Float16*)(ws + off); off = align256(off + (size_t)576 * 192 * 2);
    _Float16* fc1wT = (_Float16*)(ws + off); off = align256(off + (size_t)384 * 192 * 2);
    _Float16* fc2wT = (_Float16*)(ws + off); off = align256(off + (size_t)192 * 384 * 2);
    float*    biasP = (float*)   (ws + off); off = align256(off + (size_t)HEADS * 64 * KPAD * 4);
    _Float16* xn    = (_Float16*)(ws + off); off = align256(off + (size_t)NPIX * CDIM * 2);   // reused as h1
    _Float16* qkvH  = (_Float16*)(ws + off); off = align256(off + (size_t)NPIX * 576 * 2);    // reused as hmlp
    float*    x1    = (float*)   (ws + off); off = align256(off + (size_t)NPIX * CDIM * 4);

    // 1) weights -> f16 transposed, padded/prescaled bias table gather
    prep_kernel<<<1248, 256, 0, stream>>>(qkvw, fc1w, fc2w, rpb, rpi, qkvwT, fc1wT, fc2wT, biasP);
    // 2) LN1 -> f16
    ln_kernel<<<NPIX / 8, 256, 0, stream>>>(x, n1g, n1b, xn, NPIX);
    // 3) QKV GEMM: [NPIX,192] @ [192,576]
    gemm_kernel<0><<<9216, 256, 0, stream>>>(xn, qkvwT, qkvb, qkvH, nullptr, nullptr, NPIX, 576, CDIM);
    // 4) attention + shortcut residual -> x1 (f32)
    attn_kernel<<<NWIN * HEADS, 128, 0, stream>>>(qkvH, x, biasP, x1);
    // 5) LN2 -> f16 (reuse xn buffer)
    ln_kernel<<<NPIX / 8, 256, 0, stream>>>(x1, n2g, n2b, xn, NPIX);
    // 6) fc1 + exact GELU -> f16 (reuse qkv buffer as hmlp)
    gemm_kernel<1><<<6144, 256, 0, stream>>>(xn, fc1wT, fc1b, qkvH, nullptr, nullptr, NPIX, MLPH, CDIM);
    // 7) fc2 + bias + residual -> d_out (f32)
    gemm_kernel<2><<<3072, 256, 0, stream>>>(qkvH, fc2wT, fc2b, nullptr, (float*)d_out, x1, NPIX, CDIM, MLPH);
}
